// MultiHeadAttention_20950850470166
// MI455X (gfx1250) — compile-verified, matrix operand order checked by reading, not run on hardware
//
#include <hip/hip_runtime.h>
#include <hip/hip_bf16.h>

typedef __attribute__((ext_vector_type(16))) _Float16     v16h;
typedef __attribute__((ext_vector_type(8)))  float        v8f;
typedef __attribute__((ext_vector_type(4)))  unsigned int u32x4;

#define D_EMB 1024
#define NH    16
#define DH    64
#define BB    2
#define SS    2048
#define MM    (BB * SS)   // 4096 rows total

union Frag {
    v16h  h;
    u32x4 u[2];
};

__device__ __forceinline__ v8f wmma_f16(const Frag& a, const Frag& b, v8f c) {
    // v_wmma_f32_16x16x32_f16  D = A(16x32) * B(32x16) + C
    return __builtin_amdgcn_wmma_f32_16x16x32_f16(
        /*neg_a=*/false, a.h, /*neg_b=*/false, b.h,
        /*c_mod=*/(short)0, c, /*reuse_a=*/false, /*reuse_b=*/false);
}

// ---------------------------------------------------------------------------
// fp32 -> fp16 conversion (vectorized x4)
// ---------------------------------------------------------------------------
__global__ __launch_bounds__(256) void cvt_f32_f16(const float* __restrict__ src,
                                                   _Float16* __restrict__ dst,
                                                   int n4) {
    int i = blockIdx.x * 256 + threadIdx.x;
    if (i < n4) {
        const float4 v = reinterpret_cast<const float4*>(src)[i];
        union { _Float16 h[4]; unsigned long long q; } o;
        o.h[0] = (_Float16)v.x;
        o.h[1] = (_Float16)v.y;
        o.h[2] = (_Float16)v.z;
        o.h[3] = (_Float16)v.w;
        reinterpret_cast<unsigned long long*>(dst)[i] = o.q;
    }
}

// ---------------------------------------------------------------------------
// WMMA GEMM:  Y[M,N] = X[M,K] @ W[N,K]^T + bias   (nn.Linear semantics)
// Block = 256 threads = 8 waves -> 128x128 output tile.
// Wave tile = 32x64: two A row-tiles share four B fragments (8 WMMA / k-step),
// ~21 FLOP per cache byte vs ~13 for a 16x64 tile.
// mode 0: f16 out [M,N]
// mode 1: f16 out transposed per-head [B,H,DH,S] (for V)
// mode 2: f32 out [M,N] (final output)
// ---------------------------------------------------------------------------
__global__ __launch_bounds__(256) void gemm_wmma(
    const _Float16* __restrict__ X,   // [M,K]
    const _Float16* __restrict__ W,   // [N,K]
    const float*    __restrict__ bias,// [N]
    void*           __restrict__ Yout,
    int M, int N, int K, int mode)
{
    const int lane = threadIdx.x & 31;
    const int w    = threadIdx.x >> 5;
    const int mtiles = M >> 7;               // blocks of 128 rows
    const int bm = blockIdx.x % mtiles;
    const int bn = blockIdx.x / mtiles;
    const int m0 = bm * 128 + (w & 3) * 32;  // wave: 32 rows
    const int n0 = bn * 128 + (w >> 2) * 64; // wave: 64 cols

    const int hl  = lane >> 4;   // half-wave: selects K sub-chunk
    const int l16 = lane & 15;   // row (A) / col (B) within tile

    const _Float16* xrow = X + (size_t)(m0 + l16) * K;
    const _Float16* wrow = W + (size_t)(n0 + l16) * K;

    v8f c00 = {}, c01 = {}, c02 = {}, c03 = {};
    v8f c10 = {}, c11 = {}, c12 = {}, c13 = {};

    for (int k0 = 0; k0 < K; k0 += 32) {
        const int ka = k0 + hl * 8;
        Frag a0, a1;
        a0.u[0] = *reinterpret_cast<const u32x4*>(xrow + ka);
        a0.u[1] = *reinterpret_cast<const u32x4*>(xrow + ka + 16);
        a1.u[0] = *reinterpret_cast<const u32x4*>(xrow + 16 * K + ka);
        a1.u[1] = *reinterpret_cast<const u32x4*>(xrow + 16 * K + ka + 16);
        __builtin_prefetch(xrow + ka + 32, 0, 3);   // global_prefetch_b8, near-cache

        Frag b0, b1, b2, b3;
        b0.u[0] = *reinterpret_cast<const u32x4*>(wrow + ka);
        b0.u[1] = *reinterpret_cast<const u32x4*>(wrow + ka + 16);
        b1.u[0] = *reinterpret_cast<const u32x4*>(wrow + 16 * K + ka);
        b1.u[1] = *reinterpret_cast<const u32x4*>(wrow + 16 * K + ka + 16);
        b2.u[0] = *reinterpret_cast<const u32x4*>(wrow + 32 * K + ka);
        b2.u[1] = *reinterpret_cast<const u32x4*>(wrow + 32 * K + ka + 16);
        b3.u[0] = *reinterpret_cast<const u32x4*>(wrow + 48 * K + ka);
        b3.u[1] = *reinterpret_cast<const u32x4*>(wrow + 48 * K + ka + 16);

        c00 = wmma_f16(a0, b0, c00);
        c01 = wmma_f16(a0, b1, c01);
        c02 = wmma_f16(a0, b2, c02);
        c03 = wmma_f16(a0, b3, c03);
        c10 = wmma_f16(a1, b0, c10);
        c11 = wmma_f16(a1, b1, c11);
        c12 = wmma_f16(a1, b2, c12);
        c13 = wmma_f16(a1, b3, c13);
    }

    v8f cc[2][4] = {{c00, c01, c02, c03}, {c10, c11, c12, c13}};
    #pragma unroll
    for (int i = 0; i < 2; ++i) {
        #pragma unroll
        for (int t = 0; t < 4; ++t) {
            const int n  = n0 + t * 16 + l16;
            const float bv = bias[n];
            if (mode == 2) {
                float* Y = (float*)Yout;
                #pragma unroll
                for (int r = 0; r < 8; ++r) {
                    int row = m0 + i * 16 + r + hl * 8;
                    Y[(size_t)row * N + n] = cc[i][t][r] + bv;
                }
            } else if (mode == 0) {
                _Float16* Y = (_Float16*)Yout;
                #pragma unroll
                for (int r = 0; r < 8; ++r) {
                    int row = m0 + i * 16 + r + hl * 8;
                    Y[(size_t)row * N + n] = (_Float16)(cc[i][t][r] + bv);
                }
            } else { // mode 1: V stored transposed [B, H, DH, S]
                _Float16* Y = (_Float16*)Yout;
                const int hh = n >> 6;
                const int d  = n & (DH - 1);
                #pragma unroll
                for (int r = 0; r < 8; ++r) {
                    int row = m0 + i * 16 + r + hl * 8;
                    int bbx = row >> 11;        // / SS
                    int s   = row & (SS - 1);
                    Y[(((size_t)bbx * NH + hh) * DH + d) * SS + s] =
                        (_Float16)(cc[i][t][r] + bv);
                }
            }
        }
    }
}

// ---------------------------------------------------------------------------
// Flash attention: one wave owns a 16-row Q tile of one (b,h).
// Streams 32-wide KV blocks up to the causal frontier; online softmax.
// Q,K: f16 [B,S,D]; V: f16 [B,H,DH,S] (pre-transposed); out ctx: f16 [B,S,D].
// ---------------------------------------------------------------------------
__global__ __launch_bounds__(256) void flash_attn_wmma(
    const _Float16* __restrict__ Q,
    const _Float16* __restrict__ Km,
    const _Float16* __restrict__ Vt,
    _Float16*       __restrict__ C)
{
    __shared__ _Float16 pbuf[8][16 * 32];   // 1KB per wave P-transpose staging

    const int lane = threadIdx.x & 31;
    const int w    = threadIdx.x >> 5;
    const int wid  = blockIdx.x * 8 + w;
    const int qt   = wid & 127;             // S/16 = 128 q-tiles
    const int hh   = (wid >> 7) & (NH - 1);
    const int bb   = wid >> 11;
    const int hl   = lane >> 4;
    const int l16  = lane & 15;

    // Q tile in A layout: two 16x32 fragments over d
    const _Float16* Qb = Q + ((size_t)(bb * SS + qt * 16 + l16)) * D_EMB + hh * DH;
    Frag a0, a1;
    a0.u[0] = *(const u32x4*)(Qb + hl * 8);
    a0.u[1] = *(const u32x4*)(Qb + hl * 8 + 16);
    a1.u[0] = *(const u32x4*)(Qb + 32 + hl * 8);
    a1.u[1] = *(const u32x4*)(Qb + 32 + hl * 8 + 16);

    v8f acc0 = {}, acc1 = {}, acc2 = {}, acc3 = {};
    float mrow[8], lrow[8];
    #pragma unroll
    for (int r = 0; r < 8; ++r) { mrow[r] = -3.0e30f; lrow[r] = 0.0f; }

    const _Float16* Kb = Km + ((size_t)bb * SS) * D_EMB + hh * DH;
    const _Float16* Vb = Vt + (((size_t)bb * NH + hh) * DH) * SS;
    _Float16* pslice = &pbuf[w][0];

    const int nblk = (qt * 16 + 15) / 32 + 1;   // causal frontier

    for (int j = 0; j < nblk; ++j) {
        const int kv = j * 32;

        // ---- logits = Q K^T for 32 kv positions (two 16x16 tiles) ----
        const _Float16* Kr0 = Kb + (size_t)(kv + l16) * D_EMB;
        const _Float16* Kr1 = Kr0 + (size_t)16 * D_EMB;
        Frag kb;
        v8f s0 = {}, s1 = {};
        kb.u[0] = *(const u32x4*)(Kr0 + hl * 8);
        kb.u[1] = *(const u32x4*)(Kr0 + hl * 8 + 16);
        s0 = wmma_f16(a0, kb, s0);
        kb.u[0] = *(const u32x4*)(Kr0 + 32 + hl * 8);
        kb.u[1] = *(const u32x4*)(Kr0 + 32 + hl * 8 + 16);
        s0 = wmma_f16(a1, kb, s0);
        kb.u[0] = *(const u32x4*)(Kr1 + hl * 8);
        kb.u[1] = *(const u32x4*)(Kr1 + hl * 8 + 16);
        s1 = wmma_f16(a0, kb, s1);
        kb.u[0] = *(const u32x4*)(Kr1 + 32 + hl * 8);
        kb.u[1] = *(const u32x4*)(Kr1 + 32 + hl * 8 + 16);
        s1 = wmma_f16(a1, kb, s1);

        // ---- scale + causal mask (diagonal blocks only) ----
        const bool edge = (kv + 31) > (qt * 16);
        float p0[8], p1[8], bmax[8];
        #pragma unroll
        for (int r = 0; r < 8; ++r) {
            float x0 = s0[r] * 0.125f;   // 1/sqrt(64)
            float x1 = s1[r] * 0.125f;
            if (edge) {
                int mg = qt * 16 + r + hl * 8;
                if (kv + l16 > mg)      x0 = -3.0e30f;
                if (kv + 16 + l16 > mg) x1 = -3.0e30f;
            }
            p0[r] = x0; p1[r] = x1;
            bmax[r] = fmaxf(x0, x1);
        }

        // ---- row max across the 16 lanes holding one row ----
        #pragma unroll
        for (int r = 0; r < 8; ++r) {
            float v = bmax[r];
            v = fmaxf(v, __shfl_xor(v, 1, 32));
            v = fmaxf(v, __shfl_xor(v, 2, 32));
            v = fmaxf(v, __shfl_xor(v, 4, 32));
            v = fmaxf(v, __shfl_xor(v, 8, 32));
            bmax[r] = v;
        }

        // ---- online softmax update ----
        float sc[8];
        #pragma unroll
        for (int r = 0; r < 8; ++r) {
            float mn = fmaxf(mrow[r], bmax[r]);
            sc[r]   = __expf(mrow[r] - mn);
            mrow[r] = mn;
            p0[r] = __expf(p0[r] - mn);
            p1[r] = __expf(p1[r] - mn);
            float rs = p0[r] + p1[r];
            rs += __shfl_xor(rs, 1, 32);
            rs += __shfl_xor(rs, 2, 32);
            rs += __shfl_xor(rs, 4, 32);
            rs += __shfl_xor(rs, 8, 32);
            lrow[r] = lrow[r] * sc[r] + rs;
        }
        #pragma unroll
        for (int r = 0; r < 8; ++r) {
            acc0[r] *= sc[r]; acc1[r] *= sc[r];
            acc2[r] *= sc[r]; acc3[r] *= sc[r];
        }

        // ---- P: C-layout f32 -> A-layout f16 via per-wave LDS slice ----
        #pragma unroll
        for (int r = 0; r < 8; ++r) {
            int mrw = r + hl * 8;
            pslice[mrw * 32 + l16]      = (_Float16)p0[r];
            pslice[mrw * 32 + 16 + l16] = (_Float16)p1[r];
        }
        asm volatile("s_wait_dscnt 0" ::: "memory");
        Frag pa;
        const _Float16* pr = pslice + l16 * 32;
        pa.u[0] = *(const u32x4*)(pr + hl * 8);
        pa.u[1] = *(const u32x4*)(pr + hl * 8 + 16);

        // ---- ctx += P @ V   (V rows contiguous over kv thanks to transpose) ----
        const _Float16* Vr = Vb + (size_t)l16 * SS + kv + hl * 8;
        Frag vbf;
        vbf.u[0] = *(const u32x4*)(Vr);
        vbf.u[1] = *(const u32x4*)(Vr + 16);
        acc0 = wmma_f16(pa, vbf, acc0);
        vbf.u[0] = *(const u32x4*)(Vr + (size_t)16 * SS);
        vbf.u[1] = *(const u32x4*)(Vr + (size_t)16 * SS + 16);
        acc1 = wmma_f16(pa, vbf, acc1);
        vbf.u[0] = *(const u32x4*)(Vr + (size_t)32 * SS);
        vbf.u[1] = *(const u32x4*)(Vr + (size_t)32 * SS + 16);
        acc2 = wmma_f16(pa, vbf, acc2);
        vbf.u[0] = *(const u32x4*)(Vr + (size_t)48 * SS);
        vbf.u[1] = *(const u32x4*)(Vr + (size_t)48 * SS + 16);
        acc3 = wmma_f16(pa, vbf, acc3);
    }

    // ---- normalize and store ctx as f16 [B,S,D] ----
    #pragma unroll
    for (int r = 0; r < 8; ++r) {
        float inv = 1.0f / lrow[r];
        int row = bb * SS + qt * 16 + r + hl * 8;
        size_t base = (size_t)row * D_EMB + hh * DH + l16;
        C[base]      = (_Float16)(acc0[r] * inv);
        C[base + 16] = (_Float16)(acc1[r] * inv);
        C[base + 32] = (_Float16)(acc2[r] * inv);
        C[base + 48] = (_Float16)(acc3[r] * inv);
    }
}

// ---------------------------------------------------------------------------
extern "C" void kernel_launch(void* const* d_in, const int* in_sizes, int n_in,
                              void* d_out, int out_size, void* d_ws, size_t ws_size,
                              hipStream_t stream) {
    const float* queries = (const float*)d_in[0];
    const float* values  = (const float*)d_in[1];
    // d_in[2] = mask: causal, applied analytically in the kernel
    const float* Wq = (const float*)d_in[3];
    const float* bq = (const float*)d_in[4];
    const float* Wk = (const float*)d_in[5];
    const float* bk = (const float*)d_in[6];
    const float* Wv = (const float*)d_in[7];
    const float* bv = (const float*)d_in[8];
    const float* Wo = (const float*)d_in[9];
    const float* bo = (const float*)d_in[10];
    float* out = (float*)d_out;

    const size_t nBSD = (size_t)BB * SS * D_EMB;   // 4,194,304
    const size_t nDD  = (size_t)D_EMB * D_EMB;     // 1,048,576

    _Float16* ws  = (_Float16*)d_ws;
    _Float16* Xq  = ws; ws += nBSD;
    _Float16* Xv  = ws; ws += nBSD;
    _Float16* Wqh = ws; ws += nDD;
    _Float16* Wkh = ws; ws += nDD;
    _Float16* Wvh = ws; ws += nDD;
    _Float16* Woh = ws; ws += nDD;
    _Float16* Qh  = ws; ws += nBSD;
    _Float16* Kh  = ws; ws += nBSD;
    _Float16* Vth = ws; ws += nBSD;
    _Float16* Ch  = ws; ws += nBSD;

    // f32 -> f16 casts (weights + activations)
    {
        int n4 = (int)(nBSD / 4);
        cvt_f32_f16<<<(n4 + 255) / 256, 256, 0, stream>>>(queries, Xq, n4);
        cvt_f32_f16<<<(n4 + 255) / 256, 256, 0, stream>>>(values,  Xv, n4);
        int m4 = (int)(nDD / 4);
        cvt_f32_f16<<<(m4 + 255) / 256, 256, 0, stream>>>(Wq, Wqh, m4);
        cvt_f32_f16<<<(m4 + 255) / 256, 256, 0, stream>>>(Wk, Wkh, m4);
        cvt_f32_f16<<<(m4 + 255) / 256, 256, 0, stream>>>(Wv, Wvh, m4);
        cvt_f32_f16<<<(m4 + 255) / 256, 256, 0, stream>>>(Wo, Woh, m4);
    }

    const int M = MM, N = D_EMB, K = D_EMB;
    dim3 gblk((M / 128) * (N / 128));   // 256 workgroups

    gemm_wmma<<<gblk, 256, 0, stream>>>(Xq, Wqh, bq, Qh,  M, N, K, 0);
    gemm_wmma<<<gblk, 256, 0, stream>>>(Xv, Wkh, bk, Kh,  M, N, K, 0);
    gemm_wmma<<<gblk, 256, 0, stream>>>(Xv, Wvh, bv, Vth, M, N, K, 1);

    const int nwaves = BB * NH * (SS / 16);   // 4096 waves, 8 per block
    flash_attn_wmma<<<nwaves / 8, 256, 0, stream>>>(Qh, Kh, Vth, Ch);

    gemm_wmma<<<gblk, 256, 0, stream>>>(Ch, Woh, bo, out, M, N, K, 2);
}
